// TypeSpecificGATLayer_89086211654279
// MI455X (gfx1250) — matrix operand order repaired and checked by reading, not compiled, not run on hardware
//
#include <hip/hip_runtime.h>
#include <hip/hip_bf16.h>

typedef __attribute__((ext_vector_type(16))) _Float16 v16h;
typedef __attribute__((ext_vector_type(2)))  _Float16 h2;
typedef __attribute__((ext_vector_type(8)))  float    v8f;

#define N_NODES 4096
#define IN_DIM  128
#define OUT_DIM 64
#define SLOPE   0.2f

__device__ __forceinline__ float leaky(float x) { return fmaxf(x, SLOPE * x); }

// ---------------------------------------------------------------------------
// Stage 1: Wh = H @ W (fp32), store Wh transposed as f16 (WMMA B operand),
//          s_i = Wh @ a[:64], s_j = Wh @ a[64:]
// One block per node, 64 threads (one per output channel), both relations.
// ---------------------------------------------------------------------------
__global__ void __launch_bounds__(64)
gat_wh_kernel(const float* __restrict__ H,
              const float* __restrict__ W0, const float* __restrict__ W1,
              const float* __restrict__ a0, const float* __restrict__ a1,
              _Float16* __restrict__ WhT,
              float* __restrict__ s_i, float* __restrict__ s_j)
{
    __shared__ float h[IN_DIM];
    __shared__ float redI[OUT_DIM];
    __shared__ float redJ[OUT_DIM];
    const int n = blockIdx.x, t = threadIdx.x;
    h[t]      = H[n * IN_DIM + t];
    h[t + 64] = H[n * IN_DIM + 64 + t];
    __syncthreads();
    for (int rel = 0; rel < 2; ++rel) {
        const float* __restrict__ W = rel ? W1 : W0;
        const float* __restrict__ a = rel ? a1 : a0;
        float dot = 0.f;
        #pragma unroll 4
        for (int k = 0; k < IN_DIM; ++k) dot = fmaf(h[k], W[k * OUT_DIM + t], dot);
        WhT[(size_t)(rel * OUT_DIM + t) * N_NODES + n] = (_Float16)dot;
        redI[t] = dot * a[t];
        redJ[t] = dot * a[OUT_DIM + t];
        __syncthreads();
        for (int s = 32; s > 0; s >>= 1) {
            if (t < s) { redI[t] += redI[t + s]; redJ[t] += redJ[t + s]; }
            __syncthreads();
        }
        if (t == 0) { s_i[rel * N_NODES + n] = redI[0]; s_j[rel * N_NODES + n] = redJ[0]; }
        __syncthreads();
    }
}

// ---------------------------------------------------------------------------
// Stage 1b: per relation, global max of s_j (softmax shift) and precomputed
//           column factor tables E1 = exp(s_j), E2 = exp(SLOPE*s_j) in f16.
// ---------------------------------------------------------------------------
__global__ void __launch_bounds__(256)
gat_sprep_kernel(const float* __restrict__ s_j, float* __restrict__ mglob,
                 _Float16* __restrict__ E1, _Float16* __restrict__ E2)
{
    __shared__ float red[256];
    const int rel = blockIdx.x, t = threadIdx.x;
    float mx = -3.4e38f;
    for (int j = t; j < N_NODES; j += 256) mx = fmaxf(mx, s_j[rel * N_NODES + j]);
    red[t] = mx;
    __syncthreads();
    for (int s = 128; s > 0; s >>= 1) {
        if (t < s) red[t] = fmaxf(red[t], red[t + s]);
        __syncthreads();
    }
    if (t == 0) mglob[rel] = red[0];
    for (int j = t; j < N_NODES; j += 256) {
        const float sj = s_j[rel * N_NODES + j];
        E1[rel * N_NODES + j] = (_Float16)__expf(sj);
        E2[rel * N_NODES + j] = (_Float16)__expf(SLOPE * sj);
    }
}

// ---------------------------------------------------------------------------
// Stage 2: single pass over A.
//   p_ij = A_ij * exp(leaky(s_i+s_j) - m_i)
//        = A_ij * max(c1*E1[j], c2*E2[j])        (exp is monotone)
// built branchlessly with packed f16 ops straight into the WMMA A-operand
// layout; 4 WMMAs for p@Wh (N=64) + 1 WMMA vs an all-ones B tile for the
// softmax denominator. Grid: (256 row tiles, 2 k-groups, 2 relations),
// 4 waves/block, each wave owns a 512-column K slice; partials via atomics.
// ---------------------------------------------------------------------------
__global__ void __launch_bounds__(128)
gat_attn_kernel(const int* __restrict__ A0, const int* __restrict__ A1,
                const _Float16* __restrict__ WhT,
                const float* __restrict__ s_i,
                const _Float16* __restrict__ E1g, const _Float16* __restrict__ E2g,
                const float* __restrict__ mglob,
                float* __restrict__ accWS, float* __restrict__ denomWS)
{
    const int rel = blockIdx.z;
    const int* __restrict__ A = rel ? A1 : A0;
    const int lane = threadIdx.x & 31;
    const int wid  = threadIdx.x >> 5;
    const int M    = lane & 15;   // A-op row / B,C-op column within tile
    const int hi   = lane >> 4;   // half-wave K-offset selector
    const int i0   = blockIdx.x * 16;
    const int row  = i0 + M;
    const int slice = blockIdx.y * 4 + wid;          // 0..7
    const int jbeg  = slice * (N_NODES / 8);         // 512-column slice
    const int jend  = jbeg + (N_NODES / 8);

    const float si = s_i[rel * N_NODES + row];
    const float mi = leaky(si + mglob[rel]);          // exact row max (monotone leaky)
    const float c1 = __expf(si - mi);
    const float c2 = __expf(SLOPE * si - mi);
    const h2 c1p = { (_Float16)c1, (_Float16)c1 };
    const h2 c2p = { (_Float16)c2, (_Float16)c2 };

    const _Float16* __restrict__ E1 = E1g + rel * N_NODES;
    const _Float16* __restrict__ E2 = E2g + rel * N_NODES;
    const _Float16* __restrict__ Wb = WhT + (size_t)rel * OUT_DIM * N_NODES;
    const int* __restrict__ Arow = A + (size_t)row * N_NODES;

    v16h bone;
    #pragma unroll
    for (int k = 0; k < 16; ++k) bone[k] = (_Float16)1.0f;

    v8f c0 = {}, cc1 = {}, cc2 = {}, cc3 = {}, c4 = {};

    // A K order used by this kernel (contiguous, same as B): lanes<16 hold K 0..15; lanes>=16 hold K 16..31
    const int koffA = hi ? 16 : 0;
    // B K order used by this kernel (contiguous, same as A): lanes<16 hold K 0..15; lanes>=16 hold K 16..31
    const int koffB = hi ? 16 : 0;

    #pragma unroll 2
    for (int j0 = jbeg; j0 < jend; j0 += 32) {
        const int ka = j0 + koffA;
        const int4 aa = *(const int4*)(Arow + ka);
        const int4 ab = *(const int4*)(Arow + ka + 4);
        const int4 ac = *(const int4*)(Arow + ka + 8);
        const int4 ad = *(const int4*)(Arow + ka + 12);
        const uint4 u1a = *(const uint4*)(E1 + ka);        // 8 halves = 4 pairs
        const uint4 u1b = *(const uint4*)(E1 + ka + 8);
        const uint4 u2a = *(const uint4*)(E2 + ka);
        const uint4 u2b = *(const uint4*)(E2 + ka + 8);

        const unsigned av[16] = {(unsigned)aa.x, (unsigned)aa.y, (unsigned)aa.z, (unsigned)aa.w,
                                 (unsigned)ab.x, (unsigned)ab.y, (unsigned)ab.z, (unsigned)ab.w,
                                 (unsigned)ac.x, (unsigned)ac.y, (unsigned)ac.z, (unsigned)ac.w,
                                 (unsigned)ad.x, (unsigned)ad.y, (unsigned)ad.z, (unsigned)ad.w};
        const unsigned e1w[8] = {u1a.x, u1a.y, u1a.z, u1a.w, u1b.x, u1b.y, u1b.z, u1b.w};
        const unsigned e2w[8] = {u2a.x, u2a.y, u2a.z, u2a.w, u2b.x, u2b.y, u2b.z, u2b.w};

        union { unsigned u[8]; v16h h; } ap;
        #pragma unroll
        for (int q = 0; q < 8; ++q) {
            // adjacency mask for the packed pair: av in {0,1} exactly
            const unsigned t = (av[2 * q + 1] << 16) | av[2 * q];
            const unsigned m = (t << 16) - t;                 // 0xFFFF per set half
            const h2 p1 = c1p * __builtin_bit_cast(h2, e1w[q]);   // v_pk_mul_f16
            const h2 p2 = c2p * __builtin_bit_cast(h2, e2w[q]);
            const h2 pm = __builtin_elementwise_max(p1, p2);      // v_pk_max_num_f16
            ap.u[q] = __builtin_bit_cast(unsigned, pm) & m;
        }

        const int jb = j0 + koffB;
        const _Float16* bp = Wb + jb;
        const v16h b0 = *(const v16h*)(bp + (size_t)(M +  0) * N_NODES);
        const v16h b1 = *(const v16h*)(bp + (size_t)(M + 16) * N_NODES);
        const v16h b2 = *(const v16h*)(bp + (size_t)(M + 32) * N_NODES);
        const v16h b3 = *(const v16h*)(bp + (size_t)(M + 48) * N_NODES);

        c0  = __builtin_amdgcn_wmma_f32_16x16x32_f16(false, ap.h, false, b0,   (short)0, c0,  false, false);
        cc1 = __builtin_amdgcn_wmma_f32_16x16x32_f16(false, ap.h, false, b1,   (short)0, cc1, false, false);
        cc2 = __builtin_amdgcn_wmma_f32_16x16x32_f16(false, ap.h, false, b2,   (short)0, cc2, false, false);
        cc3 = __builtin_amdgcn_wmma_f32_16x16x32_f16(false, ap.h, false, b3,   (short)0, cc3, false, false);
        c4  = __builtin_amdgcn_wmma_f32_16x16x32_f16(false, ap.h, false, bone, (short)0, c4,  false, false);
    }

    // C layout: VGPR v -> row (v + hi*8), col = lane&15 within each 16-wide N tile.
    // Denominator tile c4: every column equals the rowsum -> lanes with M==0
    // (lane 0 and lane 16) carry rows i0+0..7 and i0+8..15 respectively.
    if (M == 0) {
        #pragma unroll
        for (int v = 0; v < 8; ++v)
            atomicAdd(&denomWS[rel * N_NODES + i0 + v + hi * 8], c4[v]);
    }

    float* accR = accWS + (size_t)rel * N_NODES * OUT_DIM;
    #pragma unroll
    for (int v = 0; v < 8; ++v) {
        const int orow = i0 + v + hi * 8;
        float* dst = accR + (size_t)orow * OUT_DIM + M;
        atomicAdd(dst +  0, c0[v]);
        atomicAdd(dst + 16, cc1[v]);
        atomicAdd(dst + 32, cc2[v]);
        atomicAdd(dst + 48, cc3[v]);
    }
}

// ---------------------------------------------------------------------------
// Stage 3: out = accB/denomB + accV/denomV + bias  (isolated rows -> 0)
// ---------------------------------------------------------------------------
__global__ void __launch_bounds__(256)
gat_final_kernel(const float* __restrict__ accWS, const float* __restrict__ denomWS,
                 const float* __restrict__ bias, float* __restrict__ out)
{
    const int idx = blockIdx.x * 256 + threadIdx.x;
    const int i = idx >> 6, o = idx & 63;
    float r = bias[o];
    #pragma unroll
    for (int rel = 0; rel < 2; ++rel) {
        const float d = denomWS[rel * N_NODES + i];
        const float v = accWS[(size_t)rel * N_NODES * OUT_DIM + idx];
        r += (d > 0.f) ? (v / d) : 0.f;
    }
    out[idx] = r;
}

// ---------------------------------------------------------------------------
extern "C" void kernel_launch(void* const* d_in, const int* in_sizes, int n_in,
                              void* d_out, int out_size, void* d_ws, size_t ws_size,
                              hipStream_t stream)
{
    (void)in_sizes; (void)n_in; (void)out_size; (void)ws_size;
    const float* H    = (const float*)d_in[0];
    const int*   A0   = (const int*)  d_in[1];
    const int*   A1   = (const int*)  d_in[2];
    const float* W0   = (const float*)d_in[3];
    const float* W1   = (const float*)d_in[4];
    const float* a0   = (const float*)d_in[5];
    const float* a1   = (const float*)d_in[6];
    const float* bias = (const float*)d_in[7];
    float* out = (float*)d_out;

    // workspace layout (all offsets multiples of 256B)
    char* ws = (char*)d_ws;
    size_t o = 0;
    _Float16* WhT   = (_Float16*)(ws + o); o += (size_t)2 * OUT_DIM * N_NODES * 2; // 1 MB
    float*    s_i   = (float*)   (ws + o); o += (size_t)2 * N_NODES * 4;           // 32 KB
    float*    s_j   = (float*)   (ws + o); o += (size_t)2 * N_NODES * 4;           // 32 KB
    float*    mglob = (float*)   (ws + o); o += 256;
    _Float16* E1    = (_Float16*)(ws + o); o += (size_t)2 * N_NODES * 2;           // 16 KB
    _Float16* E2    = (_Float16*)(ws + o); o += (size_t)2 * N_NODES * 2;           // 16 KB
    float*    accWS = (float*)   (ws + o); o += (size_t)2 * N_NODES * OUT_DIM * 4; // 2 MB
    float*    denomWS = (float*) (ws + o); o += (size_t)2 * N_NODES * 4;           // 32 KB

    // zero atomic accumulators (acc + denom are contiguous)
    hipMemsetAsync(accWS, 0,
                   (size_t)(2 * N_NODES * OUT_DIM + 2 * N_NODES) * sizeof(float), stream);

    gat_wh_kernel<<<N_NODES, 64, 0, stream>>>(H, W0, W1, a0, a1, WhT, s_i, s_j);
    gat_sprep_kernel<<<2, 256, 0, stream>>>(s_j, mglob, E1, E2);
    gat_attn_kernel<<<dim3(N_NODES / 16, 2, 2), 128, 0, stream>>>(
        A0, A1, WhT, s_i, E1, E2, mglob, accWS, denomWS);
    gat_final_kernel<<<(N_NODES * OUT_DIM) / 256, 256, 0, stream>>>(
        accWS, denomWS, bias, out);
}
